// MultiHeadAttention_41308995453152
// MI455X (gfx1250) — matrix operand order repaired and checked by reading, not compiled, not run on hardware
//
#include <hip/hip_runtime.h>

// ---------------------------------------------------------------- types
typedef _Float16 h8  __attribute__((ext_vector_type(8)));
typedef _Float16 h16 __attribute__((ext_vector_type(16)));
typedef float    f8  __attribute__((ext_vector_type(8)));
typedef unsigned int u32x4 __attribute__((ext_vector_type(4)));
typedef int          i32x4 __attribute__((ext_vector_type(4)));
typedef int          i32x8 __attribute__((ext_vector_type(8)));

#define B_ 4
#define T_ 2048
#define C_ 1024
#define H_ 16
#define HD_ 64

#ifndef __has_builtin
#define __has_builtin(x) 0
#endif
#if __has_builtin(__builtin_amdgcn_tensor_load_to_lds) && __has_builtin(__builtin_amdgcn_s_wait_tensorcnt)
#define USE_TDM 1
#else
#define USE_TDM 0
#endif

// Build a 16-half A/B fragment from two 16-byte (8-half) loads.
__device__ __forceinline__ h16 frag_from2(const _Float16* p0, const _Float16* p1) {
  h8 lo = *(const h8*)p0;
  h8 hi = *(const h8*)p1;
  return __builtin_shufflevector(lo, hi, 0,1,2,3,4,5,6,7,8,9,10,11,12,13,14,15);
}

__device__ __forceinline__ f8 wmma_f16(h16 a, h16 b, f8 c) {
  // (neg_a, A, neg_b, B, c_mod, C, reuse_a, reuse_b)
  return __builtin_amdgcn_wmma_f32_16x16x32_f16(false, a, false, b, (short)0, c, false, false);
}

// 16-lane (half-wave) reductions via ds_swizzle (group-of-32, xor masks keep
// the two 16-lane halves independent for xor<=8).
__device__ __forceinline__ float red16_max(float v) {
  v = fmaxf(v, __int_as_float(__builtin_amdgcn_ds_swizzle(__float_as_int(v), 0x041f)));
  v = fmaxf(v, __int_as_float(__builtin_amdgcn_ds_swizzle(__float_as_int(v), 0x081f)));
  v = fmaxf(v, __int_as_float(__builtin_amdgcn_ds_swizzle(__float_as_int(v), 0x101f)));
  v = fmaxf(v, __int_as_float(__builtin_amdgcn_ds_swizzle(__float_as_int(v), 0x201f)));
  return v;
}
__device__ __forceinline__ float red16_sum(float v) {
  v += __int_as_float(__builtin_amdgcn_ds_swizzle(__float_as_int(v), 0x041f));
  v += __int_as_float(__builtin_amdgcn_ds_swizzle(__float_as_int(v), 0x081f));
  v += __int_as_float(__builtin_amdgcn_ds_swizzle(__float_as_int(v), 0x101f));
  v += __int_as_float(__builtin_amdgcn_ds_swizzle(__float_as_int(v), 0x201f));
  return v;
}

// ---------------------------------------------------------------- convert
__global__ __launch_bounds__(256) void f32_to_f16_kernel(const float* __restrict__ in,
                                                         _Float16* __restrict__ out, int n) {
  int i = (blockIdx.x * 256 + threadIdx.x) * 4;
  if (i < n) {
    float4 v = *(const float4*)(in + i);
    out[i + 0] = (_Float16)v.x;
    out[i + 1] = (_Float16)v.y;
    out[i + 2] = (_Float16)v.z;
    out[i + 3] = (_Float16)v.w;
  }
}

// ---------------------------------------------------------------- GEMM  C = A(MxK) * B(NxK)^T
// block = 128 threads = 4 waves (2x2); each wave computes a 64x64 tile (4x4
// C fragments), K stepped by 32 -> 16 v_wmma per step.
template <bool OUT_F16>
__global__ __launch_bounds__(128) void gemm_rt_kernel(const _Float16* __restrict__ A,
                                                      const _Float16* __restrict__ Bm,
                                                      void* __restrict__ Cout,
                                                      int M, int N, int K) {
  const int lane   = threadIdx.x & 31;
  const int wave   = threadIdx.x >> 5;
  const int n16    = lane & 15;
  const int half16 = lane >> 4;
  const int row0   = blockIdx.y * 128 + (wave >> 1) * 64;
  const int col0   = blockIdx.x * 128 + (wave & 1) * 64;

  f8 c[4][4];
#pragma unroll
  for (int i = 0; i < 4; ++i)
#pragma unroll
    for (int j = 0; j < 4; ++j) c[i][j] = {};

  for (int k = 0; k < K; k += 32) {
    h16 a[4], b[4];
#pragma unroll
    for (int i = 0; i < 4; ++i) {
      // A fragment: lane half 0 -> K {0..7,16..23}, half 1 -> K {8..15,24..31}
      const _Float16* p = A + (size_t)(row0 + i * 16 + n16) * K + k + half16 * 8;
      a[i] = frag_from2(p, p + 16);
    }
#pragma unroll
    for (int j = 0; j < 4; ++j) {
      // B fragment: lane half 0 -> K {0..7,16..23}, half 1 -> K {8..15,24..31}
      const _Float16* p = Bm + (size_t)(col0 + j * 16 + n16) * K + k + half16 * 8;
      b[j] = frag_from2(p, p + 16);
    }
#pragma unroll
    for (int i = 0; i < 4; ++i)
#pragma unroll
      for (int j = 0; j < 4; ++j) c[i][j] = wmma_f16(a[i], b[j], c[i][j]);
  }

#pragma unroll
  for (int i = 0; i < 4; ++i) {
#pragma unroll
    for (int r = 0; r < 8; ++r) {
      const int row = row0 + i * 16 + r + 8 * half16;
#pragma unroll
      for (int j = 0; j < 4; ++j) {
        const int col = col0 + j * 16 + n16;
        if constexpr (OUT_F16)
          ((_Float16*)Cout)[(size_t)row * N + col] = (_Float16)c[i][j][r];
        else
          ((float*)Cout)[(size_t)row * N + col] = c[i][j][r];
      }
    }
  }
}

// ---------------------------------------------------------------- flash attention
// One 128-thread block per (b, h, 64-row Q block). 4 waves; wave w owns rows
// [w*16, w*16+16). K streamed via the Tensor Data Mover (TDM) straight into
// padded LDS; V transposed into LDS by the VALU lanes while the DMA runs.
__global__ __launch_bounds__(128) void flash_attn_kernel(const _Float16* __restrict__ qkv,
                                                         _Float16* __restrict__ y) {
  const int tid    = threadIdx.x;
  const int lane   = tid & 31;
  const int wave   = tid >> 5;
  const int n16    = lane & 15;
  const int half16 = lane >> 4;

  const int qb = blockIdx.x & 31;         // T/64 = 32 q-blocks
  const int h  = (blockIdx.x >> 5) & 15;  // head
  const int b  = blockIdx.x >> 9;         // batch

  const size_t ld = 3 * C_;
  const _Float16* Qg = qkv + (size_t)b * T_ * ld + h * HD_;
  const _Float16* Kg = Qg + C_;
  const _Float16* Vg = Qg + 2 * C_;

  __shared__ _Float16 lK[64][72];       // K tile  [key][d]   (+pad, 16B-aligned rows)
  __shared__ _Float16 lV[HD_][72];      // V tile transposed: [d][key]
  __shared__ _Float16 lP[4][16][64];    // per-wave P staging (C-frag -> A-frag reshape)

  // Q fragments for this wave's 16 rows (held for the whole K loop).
  const int qrow = qb * 64 + wave * 16 + n16;
  h16 qf[2];
  {
    const _Float16* qp = Qg + (size_t)qrow * ld;
#pragma unroll
    for (int ks = 0; ks < 2; ++ks) {
      const _Float16* p = qp + ks * 32 + half16 * 8;
      qf[ks] = frag_from2(p, p + 16);
    }
  }

  f8 o[4];
#pragma unroll
  for (int i = 0; i < 4; ++i) o[i] = {};
  float mrow[8], lrow[8];
#pragma unroll
  for (int r = 0; r < 8; ++r) { mrow[r] = -__builtin_inff(); lrow[r] = 0.f; }

#if USE_TDM
  const unsigned lds_k_base = (unsigned)(size_t)&lK[0][0];
#endif

  for (int kb = 0; kb <= qb; ++kb) {
    __syncthreads();  // all waves done reading previous lK/lV

#if USE_TDM
    if (wave == 0) {
      // Tensor DMA: 64x64 f16 tile, row stride 3072 halves, into LDS with
      // 4-DWORD padding after every 32 DWORDs (-> 72-half LDS row stride).
      const unsigned long long gaddr =
          (unsigned long long)(size_t)(Kg + (size_t)(kb * 64) * ld);
      u32x4 g0 = { 1u,                                  // count=1, user mode
                   lds_k_base,                          // lds_addr
                   (unsigned)(gaddr & 0xffffffffu),
                   (unsigned)((gaddr >> 32) & 0x01ffffffu) | 0x80000000u };  // type=2
      i32x8 g1 = { (int)((1u << 16)      // data_size = 2 bytes
                       | (1u << 20)      // pad_enable
                       | (4u << 22)      // pad_interval: 32 DWORDs
                       | (3u << 25)),    // pad_amount: 4 DWORDs (8 halves)
                   (int)(64u << 16),     // tensor_dim0[15:0] = 64
                   (int)(64u << 16),     // tensor_dim1[15:0] = 64
                   (int)(64u << 16),     // tile_dim0 = 64
                   64,                   // tile_dim1 = 64 (tile_dim2 = 0)
                   3072,                 // tensor_dim0_stride (halves)
                   0, 0 };
      i32x4 gz = { 0, 0, 0, 0 };
#if __has_include(<hip/amd_detail/amd_gfx1250_TDM.h>)
      i32x8 z8 = { 0, 0, 0, 0, 0, 0, 0, 0 };
      __builtin_amdgcn_tensor_load_to_lds(g0, g1, gz, gz, z8, 0);
#else
      __builtin_amdgcn_tensor_load_to_lds(g0, g1, gz, gz, 0);
#endif
    }
#endif

    // V tile: load rows and transpose into lV while the TDM streams K.
    for (int cidx = tid; cidx < 64 * 8; cidx += 128) {
      const int kk  = cidx >> 3;
      const int col = (cidx & 7) * 8;
      const size_t grow = (size_t)(kb * 64 + kk) * ld;
      union { uint4 u; _Float16 hx[8]; } uv;
      uv.u = *(const uint4*)(Vg + grow + col);
#pragma unroll
      for (int i = 0; i < 8; ++i) lV[col + i][kk] = uv.hx[i];  // transpose on store
#if !USE_TDM
      *(uint4*)&lK[kk][col] = *(const uint4*)(Kg + grow + col);
#endif
      if (kb < qb) {  // pull next tile's lines toward L2 (global_prefetch_b8)
        __builtin_prefetch(Vg + grow + 64 * ld + col, 0, 3);
        __builtin_prefetch(Kg + grow + 64 * ld + col, 0, 3);
      }
    }
#if USE_TDM
    if (wave == 0) __builtin_amdgcn_s_wait_tensorcnt(0);
#endif
    __syncthreads();

    // S = Q * K^T  (16 rows x 64 keys for this wave), scaled by 1/sqrt(HD)
    f8 s[4];
#pragma unroll
    for (int nt = 0; nt < 4; ++nt) {
      f8 acc = {};
#pragma unroll
      for (int ks = 0; ks < 2; ++ks) {
        const _Float16* bp = &lK[nt * 16 + n16][ks * 32 + half16 * 8];
        acc = wmma_f16(qf[ks], frag_from2(bp, bp + 16), acc);
      }
      s[nt] = acc;
    }
#pragma unroll
    for (int nt = 0; nt < 4; ++nt)
#pragma unroll
      for (int r = 0; r < 8; ++r) s[nt][r] *= 0.125f;

    if (kb == qb) {  // causal mask only on the diagonal tile
#pragma unroll
      for (int nt = 0; nt < 4; ++nt) {
        const int kcol = kb * 64 + nt * 16 + n16;
#pragma unroll
        for (int r = 0; r < 8; ++r) {
          const int row = qb * 64 + wave * 16 + r + 8 * half16;
          if (kcol > row) s[nt][r] = -__builtin_inff();
        }
      }
    }

    // Online softmax update (per row r of this lane's half).
#pragma unroll
    for (int r = 0; r < 8; ++r) {
      float bm = fmaxf(fmaxf(s[0][r], s[1][r]), fmaxf(s[2][r], s[3][r]));
      bm = red16_max(bm);
      const float mnew  = fmaxf(mrow[r], bm);
      const float alpha = __expf(mrow[r] - mnew);
      mrow[r] = mnew;
      float psum = 0.f;
#pragma unroll
      for (int nt = 0; nt < 4; ++nt) {
        const float p = __expf(s[nt][r] - mnew);
        s[nt][r] = p;
        psum += p;
      }
      psum   = red16_sum(psum);
      lrow[r] = lrow[r] * alpha + psum;
#pragma unroll
      for (int dt = 0; dt < 4; ++dt) o[dt][r] *= alpha;
    }

    // Stage P (C-fragment layout) to LDS, reload as A fragments.
#pragma unroll
    for (int nt = 0; nt < 4; ++nt)
#pragma unroll
      for (int r = 0; r < 8; ++r)
        lP[wave][r + 8 * half16][nt * 16 + n16] = (_Float16)s[nt][r];
    asm volatile("s_wait_dscnt 0" ::: "memory");  // wave-local LDS RAW fence

    // O += P * V   (contraction over 64 keys, 2 k-steps of 32)
#pragma unroll
    for (int ks = 0; ks < 2; ++ks) {
      const _Float16* ap = &lP[wave][n16][ks * 32 + half16 * 8];
      const h16 af = frag_from2(ap, ap + 16);
#pragma unroll
      for (int dt = 0; dt < 4; ++dt) {
        const _Float16* bp = &lV[dt * 16 + n16][ks * 32 + half16 * 8];
        o[dt] = wmma_f16(af, frag_from2(bp, bp + 16), o[dt]);
      }
    }
  }

  // Epilogue: normalize and write y[b, row, h*64 + d] (f16).
#pragma unroll
  for (int r = 0; r < 8; ++r) {
    const float inv = 1.f / lrow[r];
    const int row   = qb * 64 + wave * 16 + r + 8 * half16;
    _Float16* yp = y + ((size_t)b * T_ + row) * C_ + h * HD_;
#pragma unroll
    for (int dt = 0; dt < 4; ++dt)
      yp[dt * 16 + n16] = (_Float16)(o[dt][r] * inv);
  }
}

// ---------------------------------------------------------------- launch
extern "C" void kernel_launch(void* const* d_in, const int* in_sizes, int n_in,
                              void* d_out, int out_size, void* d_ws, size_t ws_size,
                              hipStream_t stream) {
  (void)in_sizes; (void)n_in; (void)out_size;
  const float* x  = (const float*)d_in[0];
  const float* Wa = (const float*)d_in[1];  // [3C, C]
  const float* Wp = (const float*)d_in[2];  // [C, C]
  float* out = (float*)d_out;               // [B, T, C] f32

  const size_t nX  = (size_t)B_ * T_ * C_;      // 8388608
  const size_t nWa = (size_t)3 * C_ * C_;       // 3145728
  const size_t nWp = (size_t)C_ * C_;           // 1048576
  const size_t nQKV = (size_t)B_ * T_ * 3 * C_; // 25165824

  char* ws = (char*)d_ws;
  _Float16* xh   = (_Float16*)ws;            ws += nX   * 2;
  _Float16* wah  = (_Float16*)ws;            ws += nWa  * 2;
  _Float16* wph  = (_Float16*)ws;            ws += nWp  * 2;
  _Float16* qkvh = (_Float16*)ws;            ws += nQKV * 2;
  _Float16* yh   = (_Float16*)ws;            ws += nX   * 2;
  if (ws_size < (size_t)(ws - (char*)d_ws)) return;  // ~92.3 MB needed

  f32_to_f16_kernel<<<(int)(nX  / 1024), 256, 0, stream>>>(x,  xh,  (int)nX);
  f32_to_f16_kernel<<<(int)(nWa / 1024), 256, 0, stream>>>(Wa, wah, (int)nWa);
  f32_to_f16_kernel<<<(int)(nWp / 1024), 256, 0, stream>>>(Wp, wph, (int)nWp);

  // qkv = x @ W_attn^T : M=8192, N=3072, K=1024
  dim3 g1(3072 / 128, 8192 / 128);
  gemm_rt_kernel<true><<<g1, 128, 0, stream>>>(xh, wah, (void*)qkvh, 8192, 3072, 1024);

  // flash attention: one block per (b, h, 64-row q block)
  flash_attn_kernel<<<B_ * H_ * (T_ / 64), 128, 0, stream>>>(qkvh, yh);

  // out = y @ W_proj^T : M=8192, N=1024, K=1024 (f32 out)
  dim3 g2(1024 / 128, 8192 / 128);
  gemm_rt_kernel<false><<<g2, 128, 0, stream>>>(yh, wph, (void*)out, 8192, 1024, 1024);
}
